// RectB_55834574848618
// MI455X (gfx1250) — compile-verified, matrix operand order checked by reading, not run on hardware
//
#include <hip/hip_runtime.h>
#include <hip/hip_bf16.h>

typedef __attribute__((ext_vector_type(16))) _Float16 v16h;
typedef __attribute__((ext_vector_type(8)))  _Float16 v8h;
typedef __attribute__((ext_vector_type(4)))  _Float16 v4h;
typedef __attribute__((ext_vector_type(8)))  float    v8f;
typedef __attribute__((ext_vector_type(4)))  float    v4f;

#define B_   2
#define C_   64
#define H_   256
#define W_   256
#define NPIX (B_ * H_ * W_)   // 131072
#define NPTS 9

#define PIXB 128              // pixels per block in GEMM kernels
#define ASTR 72               // padded A-row stride in halves (64 + 8 pad)

// ---------------------------------------------------------------------------
// K0: NCHW -> NHWC transpose (LDS tiled, both directions coalesced)
// grid = B*H*(W/64) = 2048, block = 256
// ---------------------------------------------------------------------------
__global__ __launch_bounds__(256)
void k_nchw2nhwc(const float* __restrict__ x, float* __restrict__ xt) {
  __shared__ float t[64][65];
  int bid = blockIdx.x;
  int wt = bid & 3;
  int h  = (bid >> 2) & 255;
  int b  = bid >> 10;
  int w0 = wt * 64;
  int tid = threadIdx.x;
#pragma unroll
  for (int i = 0; i < 16; ++i) {
    int idx = i * 256 + tid;
    int w = idx & 63, c = idx >> 6;                    // coalesced over w
    t[w][c] = x[(size_t)((b * 64 + c) * 256 + h) * 256 + w0 + w];
  }
  __syncthreads();
#pragma unroll
  for (int i = 0; i < 16; ++i) {
    int idx = i * 256 + tid;
    int c = idx & 63, w = idx >> 6;                    // coalesced over c
    xt[(size_t)((b * 256 + h) * 256 + w0 + w) * 64 + c] = t[w][c];
  }
}

// ---------------------------------------------------------------------------
// K1: pack conv weights [O][64][9] (fp32, O<=NT*16) into the per-lane f16
// B-fragment layout of v_wmma_f32_16x16x32_f16 (N padded with zeros).
//   idx = ((((n*2+s)*NT + t)*32 + lane)*16 + e)
//   N = t*16+(lane&15); K = (lane&16?8:0)+(e<8?e:e+8); c = s*32+K
// ---------------------------------------------------------------------------
template <int NT, int O>
__global__ __launch_bounds__(256)
void k_pack_w(const float* __restrict__ w, _Float16* __restrict__ wp) {
  int idx = blockIdx.x * 256 + threadIdx.x;
  if (idx >= NPTS * 2 * NT * 512) return;
  int e    = idx & 15;
  int lane = (idx >> 4) & 31;
  int t    = (idx >> 9) % NT;
  int s    = (idx / (NT * 512)) & 1;
  int n    = idx / (NT * 1024);
  int N    = t * 16 + (lane & 15);
  int K    = ((lane & 16) ? 8 : 0) + (e < 8 ? e : e + 8);
  int c    = s * 32 + K;
  float v  = (N < O) ? w[(N * 64 + c) * 9 + n] : 0.f;
  wp[idx]  = (_Float16)v;
}

// ---------------------------------------------------------------------------
// K2: offset-field conv as an im2col WMMA GEMM.
//   out[pix][18] = bias + sum_{c,n} src[p+pn][c] * w_off[j][c][n]
//   M = 128 pixels/block, N = 32 (18 used), K = 576 (9 chunks of 64).
//   Double-buffered f16 A-chunks in LDS; one barrier per chunk.
// grid = NPIX/128 = 1024, block = 256 (8 waves)
// ---------------------------------------------------------------------------
__global__ __launch_bounds__(256)
void k_off_gemm(const float* __restrict__ src, const _Float16* __restrict__ wp,
                const float* __restrict__ bo, float* __restrict__ off) {
  __shared__ __align__(16) _Float16 A[2][PIXB * ASTR];   // 36 KB
  __shared__ int   sI[NPTS][PIXB];
  __shared__ float sM[NPTS][PIXB];

  const int tid  = threadIdx.x;
  const int lane = tid & 31;
  const int wv   = tid >> 5;
  const int pix0 = blockIdx.x * PIXB;
  const int ncol = lane & 15;

  v8f acc[2];
#pragma unroll
  for (int t = 0; t < 2; ++t) {
    int o = t * 16 + ncol;
    float bz = (o < 18) ? bo[o] : 0.f;
#pragma unroll
    for (int r = 0; r < 8; ++r) acc[t][r] = bz;
  }

  // im2col tap addresses (float4 units) + in-bounds masks, all 9 taps
  for (int i = tid; i < NPTS * PIXB; i += 256) {
    int n = i >> 7, pl = i & 127;
    int p = pix0 + pl;
    int b = p >> 16, h = (p >> 8) & 255, w = p & 255;
    int hh = h + n / 3 - 1, ww = w + n % 3 - 1;
    bool inb = (unsigned)hh < 256u && (unsigned)ww < 256u;
    sI[n][pl] = inb ? (b * 65536 + hh * 256 + ww) * 16 : 0;
    sM[n][pl] = inb ? 1.f : 0.f;
  }
  __syncthreads();

  const v4f* s4 = (const v4f*)src;
  for (int n = 0; n < NPTS; ++n) {
    int pb = n & 1;
    // build A-chunk [128][64] f16 via float4 loads
#pragma unroll
    for (int k = 0; k < 8; ++k) {
      int idx = k * 256 + tid;
      int pl = idx >> 4, g = idx & 15;
      v4f v = sM[n][pl] * s4[sI[n][pl] + g];
      *(v4h*)&A[pb][pl * ASTR + g * 4] = __builtin_convertvector(v, v4h);
    }
    __syncthreads();
#pragma unroll
    for (int s = 0; s < 2; ++s) {
      int arow  = wv * 16 + (lane & 15);
      int abase = s * 32 + ((lane & 16) ? 8 : 0);
      const v8h* ap = (const v8h*)&A[pb][arow * ASTR + abase];
      v16h af = __builtin_shufflevector(ap[0], ap[2],
                 0,1,2,3,4,5,6,7,8,9,10,11,12,13,14,15);
      const _Float16* bb = wp + ((size_t)(n * 2 + s) * 2 * 32 + lane) * 16;
#pragma unroll
      for (int t = 0; t < 2; ++t) {
        v16h bf = *(const v16h*)(bb + t * 512);
        acc[t] = __builtin_amdgcn_wmma_f32_16x16x32_f16(
                     false, af, false, bf, (short)0, acc[t], false, false);
      }
    }
  }

  int mrow = pix0 + wv * 16 + ((lane >> 4) << 3);
#pragma unroll
  for (int t = 0; t < 2; ++t) {
    int o = t * 16 + ncol;
    if (o < 18) {
#pragma unroll
      for (int r = 0; r < 8; ++r)
        off[(size_t)(mrow + r) * 18 + o] = acc[t][r];
    }
  }
}

// ---------------------------------------------------------------------------
// K3: fused bilinear sampling + WMMA contraction.
//   M = 128 pixels/block, N = 64 (4 tiles), K = 576 (9 chunks of 64).
//   All bilinear params precomputed once; double-buffered A; float4 taps.
// grid = NPIX/128 = 1024, block = 256 (8 waves)
// ---------------------------------------------------------------------------
template <bool RELU>
__global__ __launch_bounds__(256)
void k_deform_gemm(const float* __restrict__ src, const float* __restrict__ off,
                   const _Float16* __restrict__ wp, float* __restrict__ dst) {
  __shared__ __align__(16) _Float16 A[2][PIXB * ASTR];   // 36 KB
  __shared__ int   sI[NPTS][4][PIXB];                    // 18 KB
  __shared__ float sG[NPTS][4][PIXB];                    // 18 KB

  const int tid  = threadIdx.x;
  const int lane = tid & 31;
  const int wv   = tid >> 5;
  const int pix0 = blockIdx.x * PIXB;

  v8f acc[4] = {v8f{}, v8f{}, v8f{}, v8f{}};

  // ---- bilinear taps + weights for all 9 sampling points ----
  for (int i = tid; i < NPTS * PIXB; i += 256) {
    int n = i >> 7, pl = i & 127;
    int p = pix0 + pl;
    int b = p >> 16, h = (p >> 8) & 255, w = p & 255;
    const float* op = off + (size_t)p * 18;
    float px = op[n]     + (float)(h + n / 3);      // h+1 + (n/3 - 1)
    float py = op[n + 9] + (float)(w + n % 3);      // w+1 + (n%3 - 1)
    float fx = floorf(px), fy = floorf(py);
    float qltx = fminf(fmaxf(fx, 0.f), 257.f);
    float qlty = fminf(fmaxf(fy, 0.f), 257.f);
    float qrbx = fminf(fmaxf(fx + 1.f, 0.f), 257.f);
    float qrby = fminf(fmaxf(fy + 1.f, 0.f), 257.f);
    float pxc  = fminf(fmaxf(px, 0.f), 257.f);
    float pyc  = fminf(fmaxf(py, 0.f), 257.f);
    float wxl = 1.f + (qltx - pxc);
    float wxr = 1.f - (qrbx - pxc);
    float wyl = 1.f + (qlty - pyc);
    float wyr = 1.f - (qrby - pyc);
    int ltx = (int)qltx - 1, lty = (int)qlty - 1;   // padded -> unpadded
    int rbx = (int)qrbx - 1, rby = (int)qrby - 1;
    int base = b * 65536;
    bool i0 = (unsigned)ltx < 256u && (unsigned)lty < 256u;  // lt
    bool i1 = (unsigned)rbx < 256u && (unsigned)rby < 256u;  // rb
    bool i2 = (unsigned)ltx < 256u && (unsigned)rby < 256u;  // lb
    bool i3 = (unsigned)rbx < 256u && (unsigned)lty < 256u;  // rt
    sI[n][0][pl] = i0 ? (base + ltx * 256 + lty) * 16 : 0;   // float4 units
    sI[n][1][pl] = i1 ? (base + rbx * 256 + rby) * 16 : 0;
    sI[n][2][pl] = i2 ? (base + ltx * 256 + rby) * 16 : 0;
    sI[n][3][pl] = i3 ? (base + rbx * 256 + lty) * 16 : 0;
    sG[n][0][pl] = i0 ? wxl * wyl : 0.f;
    sG[n][1][pl] = i1 ? wxr * wyr : 0.f;
    sG[n][2][pl] = i2 ? wxl * wyr : 0.f;
    sG[n][3][pl] = i3 ? wxr * wyl : 0.f;
  }
  __syncthreads();

  const v4f* s4 = (const v4f*)src;
  for (int n = 0; n < NPTS; ++n) {
    int pb = n & 1;
    if (n + 1 < NPTS)
      __builtin_prefetch(wp + (size_t)(n + 1) * 4096, 0, 1);
    // ---- build f16 A-chunk [128 pix][64 ch]: 4 float4 taps / thread ----
#pragma unroll
    for (int k = 0; k < 8; ++k) {
      int idx = k * 256 + tid;
      int pl = idx >> 4, g = idx & 15;
      v4f v = sG[n][0][pl] * s4[sI[n][0][pl] + g]
            + sG[n][1][pl] * s4[sI[n][1][pl] + g]
            + sG[n][2][pl] * s4[sI[n][2][pl] + g]
            + sG[n][3][pl] * s4[sI[n][3][pl] + g];
      *(v4h*)&A[pb][pl * ASTR + g * 4] = __builtin_convertvector(v, v4h);
    }
    __syncthreads();
    // ---- 2 k-steps x 4 n-tiles of v_wmma_f32_16x16x32_f16 ----
#pragma unroll
    for (int s = 0; s < 2; ++s) {
      int arow  = wv * 16 + (lane & 15);
      int abase = s * 32 + ((lane & 16) ? 8 : 0);
      const v8h* ap = (const v8h*)&A[pb][arow * ASTR + abase];
      v16h af = __builtin_shufflevector(ap[0], ap[2],
                 0,1,2,3,4,5,6,7,8,9,10,11,12,13,14,15);
      const _Float16* bb = wp + ((size_t)(n * 2 + s) * 4 * 32 + lane) * 16;
#pragma unroll
      for (int t = 0; t < 4; ++t) {
        v16h bf = *(const v16h*)(bb + t * 512);
        acc[t] = __builtin_amdgcn_wmma_f32_16x16x32_f16(
                     false, af, false, bf, (short)0, acc[t], false, false);
      }
    }
  }

  // ---- epilogue: C/D layout -> NHWC stores (two 64B segments / store) ----
  int mrow = pix0 + wv * 16 + ((lane >> 4) << 3);
  int ncol = lane & 15;
#pragma unroll
  for (int t = 0; t < 4; ++t) {
#pragma unroll
    for (int r = 0; r < 8; ++r) {
      float v = acc[t][r];
      if (RELU) v = fmaxf(v, 0.f);
      dst[(size_t)(mrow + r) * 64 + (t * 16 + ncol)] = v;
    }
  }
}

// ---------------------------------------------------------------------------
// K4: residual add + NHWC -> NCHW (LDS tiled, both sides coalesced)
// grid = 2048, block = 256
// ---------------------------------------------------------------------------
__global__ __launch_bounds__(256)
void k_add_out(const float* __restrict__ y, const float* __restrict__ x,
               float* __restrict__ out) {
  __shared__ float t[64][65];
  int bid = blockIdx.x;
  int wt = bid & 3;
  int h  = (bid >> 2) & 255;
  int b  = bid >> 10;
  int w0 = wt * 64;
  int tid = threadIdx.x;
#pragma unroll
  for (int i = 0; i < 16; ++i) {
    int idx = i * 256 + tid;
    int c = idx & 63, w = idx >> 6;                    // coalesced over c
    t[w][c] = y[(size_t)((b * 256 + h) * 256 + w0 + w) * 64 + c];
  }
  __syncthreads();
#pragma unroll
  for (int i = 0; i < 16; ++i) {
    int idx = i * 256 + tid;
    int w = idx & 63, c = idx >> 6;                    // coalesced over w
    size_t oi = (size_t)((b * 64 + c) * 256 + h) * 256 + w0 + w;
    out[oi] = x[oi] + t[w][c];
  }
}

// ---------------------------------------------------------------------------
extern "C" void kernel_launch(void* const* d_in, const int* in_sizes, int n_in,
                              void* d_out, int out_size, void* d_ws, size_t ws_size,
                              hipStream_t stream) {
  (void)in_sizes; (void)n_in; (void)out_size; (void)ws_size;
  const float* x      = (const float*)d_in[0];
  const float* w_off1 = (const float*)d_in[1];
  const float* b_off1 = (const float*)d_in[2];
  const float* w1     = (const float*)d_in[3];
  const float* w_off2 = (const float*)d_in[4];
  const float* b_off2 = (const float*)d_in[5];
  const float* w2     = (const float*)d_in[6];
  float* out = (float*)d_out;

  char* ws = (char*)d_ws;
  float*    xt   = (float*)(ws);                  // 33554432 B (NHWC x; reused as y2)
  float*    y1   = (float*)(ws + 33554432);       // 33554432 B (NHWC layer-1 out)
  float*    off  = (float*)(ws + 67108864);       //  9437184 B ([pix][18], reused)
  _Float16* wp1  = (_Float16*)(ws + 76546048);    //    73728 B (main conv 1)
  _Float16* wp2  = (_Float16*)(ws + 76619776);    //    73728 B (main conv 2)
  _Float16* wpo1 = (_Float16*)(ws + 76693504);    //    36864 B (offset conv 1)
  _Float16* wpo2 = (_Float16*)(ws + 76730368);    //    36864 B (offset conv 2)
  float*    y2   = xt;                            // reuse: xt dead after layer-1 GEMM

  k_nchw2nhwc        <<<2048, 256, 0, stream>>>(x, xt);
  k_pack_w<4, 64>    <<< 144, 256, 0, stream>>>(w1, wp1);
  k_pack_w<4, 64>    <<< 144, 256, 0, stream>>>(w2, wp2);
  k_pack_w<2, 18>    <<<  72, 256, 0, stream>>>(w_off1, wpo1);
  k_pack_w<2, 18>    <<<  72, 256, 0, stream>>>(w_off2, wpo2);

  k_off_gemm         <<<1024, 256, 0, stream>>>(xt, wpo1, b_off1, off);
  k_deform_gemm<true><<<1024, 256, 0, stream>>>(xt, off, wp1, y1);

  k_off_gemm         <<<1024, 256, 0, stream>>>(y1, wpo2, b_off2, off);
  k_deform_gemm<false><<<1024, 256, 0, stream>>>(y1, off, wp2, y2);

  k_add_out          <<<2048, 256, 0, stream>>>(y2, x, out);
}
